// GatedAttention_52974126629613
// MI455X (gfx1250) — matrix operand order repaired and checked by reading, not compiled, not run on hardware
//
#include <hip/hip_runtime.h>

// ---------------------------------------------------------------------------
// GatedAttention (R-Net) on MI455X / gfx1250.
// bf16 WMMA (v_wmma_f32_16x16x32_bf16) for all GEMMs, f32 accumulation.
// TDM (tensor_load_to_lds) stages the attention 'a' tile into LDS.
// ---------------------------------------------------------------------------

typedef __bf16 bf16;
typedef __attribute__((ext_vector_type(16))) __bf16 v16bf;
typedef __attribute__((ext_vector_type(8)))  __bf16 v8bf;
typedef __attribute__((ext_vector_type(8)))  float  v8f;
typedef __attribute__((ext_vector_type(4)))  float  v4f;
typedef __attribute__((ext_vector_type(4)))  unsigned int v4u;
typedef __attribute__((ext_vector_type(8)))  int    v8i;
typedef __attribute__((ext_vector_type(4)))  int    v4i;

#define BB 64   // batch
#define PP 512  // passage length
#define QQ 64   // question length
#define DD 512  // 2*hidden
#define H4 1024
#define TH 1536 // 3*DD

#define WMMA_BF16(af, bfv, acc) \
  __builtin_amdgcn_wmma_f32_16x16x32_bf16(false, (af), false, (bfv), (short)0, (acc), false, false)

// ---- fragment loaders (layouts per CDNA5 ISA 7.12.2, wave32) --------------

// B matrix (KxN tile of W^T, W row-major NxK bf16): lane holds column
// n = lane%16; K chunks {kbase+off .. +7} and {kbase+off+16 .. +23},
// off = (lane<16 ? 0 : 8).
__device__ __forceinline__ v16bf load_b_frag(const bf16* __restrict__ W, int ldk,
                                             int n, int kbase, int lane) {
  const bf16* p = W + (size_t)n * ldk + kbase + ((lane & 16) ? 8 : 0);
  v8bf lo = *reinterpret_cast<const v8bf*>(p);
  v8bf hi = *reinterpret_cast<const v8bf*>(p + 16);
  return __builtin_shufflevector(lo, hi, 0,1,2,3,4,5,6,7,8,9,10,11,12,13,14,15);
}

// A matrix from global f32 (convert to bf16 on the fly): lane holds row
// m = lane%16; K chunks {kbase+off .. +7} and {kbase+off+16 .. +23},
// off = (lane<16 ? 0 : 8).
__device__ __forceinline__ v16bf load_a_frag_g(const float* __restrict__ A, size_t lda,
                                               int m, int kbase, int lane) {
  const float* p = A + (size_t)m * lda + kbase + ((lane & 16) ? 8 : 0);
  v4f a0 = *reinterpret_cast<const v4f*>(p);
  v4f a1 = *reinterpret_cast<const v4f*>(p + 4);
  v4f a2 = *reinterpret_cast<const v4f*>(p + 16);
  v4f a3 = *reinterpret_cast<const v4f*>(p + 20);
  v16bf r;
#pragma unroll
  for (int i = 0; i < 4; ++i) {
    r[i]      = (bf16)a0[i];
    r[4 + i]  = (bf16)a1[i];
    r[8 + i]  = (bf16)a2[i];
    r[12 + i] = (bf16)a3[i];
  }
  return r;
}

#define LDSTR 520  // padded row stride (bf16 elems) to break LDS bank conflicts

__device__ __forceinline__ v16bf load_a_frag_lds(const bf16* sm, int row,
                                                 int kbase, int lane) {
  const bf16* p = sm + row * LDSTR + kbase + ((lane & 16) ? 8 : 0);
  v8bf lo = *reinterpret_cast<const v8bf*>(p);       // K +0..7
  v8bf hi = *reinterpret_cast<const v8bf*>(p + 16);  // K +16..23
  return __builtin_shufflevector(lo, hi, 0,1,2,3,4,5,6,7,8,9,10,11,12,13,14,15);
}

// ---- generic bf16 WMMA GEMM: C[M,N] (+)= A[M,K](f32,on-the-fly bf16) @ W^T --
// grid = (N/64, M/16), block = 128 (4 waves, each one 16x16 N-subtile).
__global__ void __launch_bounds__(128)
k_gemm(const float* __restrict__ A, long lda, const bf16* __restrict__ W,
       const float* __restrict__ bias, float* __restrict__ C,
       int N, int K, int accumulate) {
  int lane = threadIdx.x & 31;
  int wave = threadIdx.x >> 5;
  int nb = blockIdx.x * 64 + wave * 16;
  int mb = blockIdx.y * 16;
  int m  = mb + (lane & 15);
  int n  = nb + (lane & 15);
  v8f acc = {0.f, 0.f, 0.f, 0.f, 0.f, 0.f, 0.f, 0.f};
  for (int kb = 0; kb < K; kb += 32) {
    __builtin_prefetch(A + (size_t)m * lda + kb + 64, 0, 1);  // global_prefetch_b8
    v16bf af  = load_a_frag_g(A, (size_t)lda, m, kb, lane);
    v16bf bfv = load_b_frag(W, K, n, kb, lane);
    acc = WMMA_BF16(af, bfv, acc);
  }
  int roff = (lane & 16) ? 8 : 0;
  float bv = bias ? bias[n] : 0.0f;
#pragma unroll
  for (int r = 0; r < 8; ++r) {
    size_t idx = (size_t)(mb + roff + r) * N + n;
    float v = acc[r] + bv;
    if (accumulate) v += C[idx];
    C[idx] = v;
  }
}

// ---- dual-source GEMM: C = A1 @ W1^T + A2 @ W2^T (register accumulate) -----
__global__ void __launch_bounds__(128)
k_gemm2(const float* __restrict__ A1, long lda1, const bf16* __restrict__ W1, int K1,
        const float* __restrict__ A2, long lda2, const bf16* __restrict__ W2, int K2,
        float* __restrict__ C, int N) {
  int lane = threadIdx.x & 31;
  int wave = threadIdx.x >> 5;
  int nb = blockIdx.x * 64 + wave * 16;
  int mb = blockIdx.y * 16;
  int m  = mb + (lane & 15);
  int n  = nb + (lane & 15);
  v8f acc = {0.f, 0.f, 0.f, 0.f, 0.f, 0.f, 0.f, 0.f};
  for (int kb = 0; kb < K1; kb += 32) {
    v16bf af  = load_a_frag_g(A1, (size_t)lda1, m, kb, lane);
    v16bf bfv = load_b_frag(W1, K1, n, kb, lane);
    acc = WMMA_BF16(af, bfv, acc);
  }
  for (int kb = 0; kb < K2; kb += 32) {
    v16bf af  = load_a_frag_g(A2, (size_t)lda2, m, kb, lane);
    v16bf bfv = load_b_frag(W2, K2, n, kb, lane);
    acc = WMMA_BF16(af, bfv, acc);
  }
  int roff = (lane & 16) ? 8 : 0;
#pragma unroll
  for (int r = 0; r < 8; ++r)
    C[(size_t)(mb + roff + r) * N + n] = acc[r];
}

// ---- fused attention GEMM: s[b,q,:] = tanh(a[b,q,:] + add[b,:]) @ Wv^T -----
// grid = (2, B): each WG owns one batch b (rows q=0..63) and half of N=512.
// Stage 0: TDM DMA of the 64x512 f32 'a' tile into LDS (wave 0 issues it).
// Stage 1: 256 threads compute tanh into LDS (bf16). Stage 2: 8 waves =
// 4 (q-tiles) x 2 (N-halves of 128), each wave 8 accumulators of 16x16.
__global__ void __launch_bounds__(256)
k_attn(const float* __restrict__ a, const float* __restrict__ addv,
       const bf16* __restrict__ Wv, float* __restrict__ s) {
  __shared__ float smf[QQ * DD];      // raw f32 'a' tile (TDM destination)
  __shared__ bf16  smb[QQ * LDSTR];   // tanh(...) in bf16, padded rows
  int b    = blockIdx.y;
  int tid  = threadIdx.x;
  int lane = tid & 31;
  int wave = tid >> 5;

  const float* abase = a + (size_t)b * QQ * DD;
  const float* ad    = addv + (size_t)b * DD;

#if __has_builtin(__builtin_amdgcn_tensor_load_to_lds) && __has_builtin(__builtin_amdgcn_s_wait_tensorcnt)
  if (wave == 0) {
    // Tensor DMA descriptor (D#), CDNA5 ISA ch.8: 2D tensor, 4-byte elems,
    // tile = full 512x64 region, contiguous in both global and LDS.
    unsigned long long gaddr = (unsigned long long)(uintptr_t)abase;
    unsigned int lds_off = (unsigned int)(uintptr_t)(&smf[0]);
    v4u g0;
    g0[0] = 1u;                                          // count=1, user desc
    g0[1] = lds_off;                                     // lds_addr (bytes)
    g0[2] = (unsigned int)(gaddr & 0xffffffffu);         // global_addr lo
    g0[3] = (unsigned int)((gaddr >> 32) & 0x1ffffffu)   // global_addr hi
            | (2u << 30);                                // type = 2 ("image")
    v8i g1;
    g1[0] = (int)(2u << 16);                             // data_size = 4B
    g1[1] = (int)((unsigned)(DD & 0xffff) << 16);        // tensor_dim0 lo16
    g1[2] = (int)((DD >> 16) | ((unsigned)(QQ & 0xffff) << 16)); // dim0 hi / dim1 lo
    g1[3] = (int)((QQ >> 16) | ((unsigned)(DD & 0xffff) << 16)); // dim1 hi / tile_dim0
    g1[4] = (int)(QQ & 0xffff);                          // tile_dim1 (tile_dim2=0)
    g1[5] = (int)DD;                                     // tensor_dim0_stride lo32
    g1[6] = 0;                                           // stride hi / dim1_stride lo
    g1[7] = 0;
    v4i g2 = {0, 0, 0, 0};
    v4i g3 = {0, 0, 0, 0};
#if defined(__clang_major__) && (__clang_major__ >= 23)
    v8i g4 = {0, 0, 0, 0, 0, 0, 0, 0};
    __builtin_amdgcn_tensor_load_to_lds(g0, g1, g2, g3, g4, 0);
#else
    __builtin_amdgcn_tensor_load_to_lds(g0, g1, g2, g3, 0);
#endif
    __builtin_amdgcn_s_wait_tensorcnt(0);
  }
  __syncthreads();
  for (int idx = tid; idx < QQ * 256; idx += 256) {
    int row = idx >> 8;
    int k0  = (idx & 255) * 2;
    float v0 = tanhf(smf[row * DD + k0]     + ad[k0]);
    float v1 = tanhf(smf[row * DD + k0 + 1] + ad[k0 + 1]);
    smb[row * LDSTR + k0]     = (bf16)v0;
    smb[row * LDSTR + k0 + 1] = (bf16)v1;
  }
#else
  for (int idx = tid; idx < QQ * 256; idx += 256) {
    int row = idx >> 8;
    int k0  = (idx & 255) * 2;
    float v0 = tanhf(abase[row * DD + k0]     + ad[k0]);
    float v1 = tanhf(abase[row * DD + k0 + 1] + ad[k0 + 1]);
    smb[row * LDSTR + k0]     = (bf16)v0;
    smb[row * LDSTR + k0 + 1] = (bf16)v1;
  }
#endif
  __syncthreads();

  int mq   = wave >> 1;            // q-tile 0..3
  int nh   = wave & 1;             // N half within this block's 256 cols
  int nbase = blockIdx.x * 256 + nh * 128;

  v8f acc[8];
#pragma unroll
  for (int j = 0; j < 8; ++j) acc[j] = (v8f){0.f,0.f,0.f,0.f,0.f,0.f,0.f,0.f};

  int arow = mq * 16 + (lane & 15);
  for (int kb = 0; kb < DD; kb += 32) {
    v16bf af = load_a_frag_lds(smb, arow, kb, lane);
#pragma unroll
    for (int j = 0; j < 8; ++j) {
      v16bf bfv = load_b_frag(Wv, DD, nbase + j * 16 + (lane & 15), kb, lane);
      acc[j] = WMMA_BF16(af, bfv, acc[j]);
    }
  }
  int roff = (lane & 16) ? 8 : 0;
#pragma unroll
  for (int j = 0; j < 8; ++j) {
#pragma unroll
    for (int r = 0; r < 8; ++r) {
      size_t row = (size_t)(b * QQ + mq * 16 + roff + r);
      s[row * DD + nbase + j * 16 + (lane & 15)] = acc[j][r];
    }
  }
}

// ---- pointwise kernels ----------------------------------------------------

__global__ void k_convert(const float* __restrict__ in, bf16* __restrict__ out, int n) {
  int i = blockIdx.x * blockDim.x + threadIdx.x;
  if (i < n) out[i] = (bf16)in[i];
}

__global__ void k_zero(float* __restrict__ p, int n) {
  int i = blockIdx.x * blockDim.x + threadIdx.x;
  if (i < n) p[i] = 0.0f;
}

// softmax over batch axis: stats per (q,d)
__global__ void k_sm_stats(const float* __restrict__ s, float* __restrict__ mbuf,
                           float* __restrict__ zbuf) {
  int t = blockIdx.x * blockDim.x + threadIdx.x;
  if (t >= QQ * DD) return;
  int q = t >> 9, d = t & 511;
  float m = -3.0e38f;
  for (int b = 0; b < BB; ++b)
    m = fmaxf(m, s[((size_t)(b * QQ + q)) * DD + d]);
  float z = 0.f;
  for (int b = 0; b < BB; ++b)
    z += __expf(s[((size_t)(b * QQ + q)) * DD + d] - m);
  mbuf[t] = m;
  zbuf[t] = 1.0f / z;
}

// c_t[b,d] = sum_q question[b,q,d] * softmax_b(s)[b,q,d]
__global__ void k_ctx(const float* __restrict__ s, const float* __restrict__ question,
                      const float* __restrict__ mbuf, const float* __restrict__ zbuf,
                      float* __restrict__ ct) {
  int t = blockIdx.x * blockDim.x + threadIdx.x;
  if (t >= BB * DD) return;
  int b = t >> 9, d = t & 511;
  float acc = 0.f;
  for (int q = 0; q < QQ; ++q) {
    size_t i = ((size_t)(b * QQ + q)) * DD + d;
    float w = __expf(s[i] - mbuf[q * DD + d]) * zbuf[q * DD + d];
    acc += w * question[i];
  }
  ct[t] = acc;
}

__global__ void k_pa(const float* __restrict__ passage, const float* __restrict__ ct,
                     float* __restrict__ pa, int t) {
  int i = blockIdx.x * blockDim.x + threadIdx.x;
  if (i >= BB * H4) return;
  int b = i >> 10, j = i & 1023;
  pa[i] = (j < DD) ? passage[(size_t)b * PP * DD + (size_t)t * DD + j]
                   : ct[b * DD + (j - DD)];
}

__global__ void k_gate(const float* __restrict__ gt_raw, const float* __restrict__ pa,
                       float* __restrict__ x) {
  int i = blockIdx.x * blockDim.x + threadIdx.x;
  if (i >= BB * H4) return;
  float g = 1.0f / (1.0f + __expf(-gt_raw[i]));
  x[i] = g * pa[i];
}

__global__ void k_gru(const float* __restrict__ gi, const float* __restrict__ gh,
                      const float* __restrict__ hprev, long hlda,
                      float* __restrict__ out, int t) {
  int i = blockIdx.x * blockDim.x + threadIdx.x;
  if (i >= BB * DD) return;
  int b = i >> 9, d = i & 511;
  float ir = gi[(size_t)b * TH + d];
  float iz = gi[(size_t)b * TH + DD + d];
  float in = gi[(size_t)b * TH + 2 * DD + d];
  float hr = gh[(size_t)b * TH + d];
  float hz = gh[(size_t)b * TH + DD + d];
  float hn = gh[(size_t)b * TH + 2 * DD + d];
  float h  = hprev[(size_t)b * hlda + d];
  float r  = 1.0f / (1.0f + __expf(-(ir + hr)));
  float z  = 1.0f / (1.0f + __expf(-(iz + hz)));
  float n  = tanhf(in + r * hn);
  out[(size_t)b * PP * DD + (size_t)t * DD + d] = (1.0f - z) * n + z * h;
}

// ---------------------------------------------------------------------------

extern "C" void kernel_launch(void* const* d_in, const int* in_sizes, int n_in,
                              void* d_out, int out_size, void* d_ws, size_t ws_size,
                              hipStream_t stream) {
  const float* passage  = (const float*)d_in[0];
  const float* question = (const float*)d_in[1];
  const float* Wuq  = (const float*)d_in[2];
  const float* Wup  = (const float*)d_in[3];
  const float* Wvp  = (const float*)d_in[4];
  const float* Wv   = (const float*)d_in[5];
  const float* Wg   = (const float*)d_in[6];
  const float* W_ih = (const float*)d_in[7];
  const float* W_hh = (const float*)d_in[8];
  const float* b_ih = (const float*)d_in[9];
  const float* b_hh = (const float*)d_in[10];
  float* out = (float*)d_out;

  // workspace carving (256B aligned)
  size_t off = 0;
  char* base = (char*)d_ws;
  auto carve = [&](size_t bytes) -> char* {
    char* p = base + off;
    off += (bytes + 255) & ~(size_t)255;
    return p;
  };
  float* a_ws   = (float*)carve((size_t)BB * QQ * DD * 4);  // 8 MB
  float* s_ws   = (float*)carve((size_t)BB * QQ * DD * 4);  // 8 MB
  float* add_ws = (float*)carve((size_t)BB * DD * 4);
  float* m_ws   = (float*)carve((size_t)QQ * DD * 4);
  float* z_ws   = (float*)carve((size_t)QQ * DD * 4);
  float* ct_ws  = (float*)carve((size_t)BB * DD * 4);
  float* pa_ws  = (float*)carve((size_t)BB * H4 * 4);
  float* gt_ws  = (float*)carve((size_t)BB * H4 * 4);
  float* x_ws   = (float*)carve((size_t)BB * H4 * 4);
  float* gi_ws  = (float*)carve((size_t)BB * TH * 4);
  float* gh_ws  = (float*)carve((size_t)BB * TH * 4);
  float* h0_ws  = (float*)carve((size_t)BB * DD * 4);
  bf16* wuq_bf  = (bf16*)carve((size_t)DD * DD * 2);
  bf16* wup_bf  = (bf16*)carve((size_t)DD * DD * 2);
  bf16* wvp_bf  = (bf16*)carve((size_t)DD * DD * 2);
  bf16* wv_bf   = (bf16*)carve((size_t)DD * DD * 2);
  bf16* wg_bf   = (bf16*)carve((size_t)H4 * H4 * 2);
  bf16* wih_bf  = (bf16*)carve((size_t)TH * H4 * 2);
  bf16* whh_bf  = (bf16*)carve((size_t)TH * DD * 2);

  // ---- one-time precompute ----
  k_zero<<<dim3((BB * DD + 255) / 256), 256, 0, stream>>>(h0_ws, BB * DD);
  k_convert<<<dim3((DD*DD + 255) / 256), 256, 0, stream>>>(Wuq,  wuq_bf, DD * DD);
  k_convert<<<dim3((DD*DD + 255) / 256), 256, 0, stream>>>(Wup,  wup_bf, DD * DD);
  k_convert<<<dim3((DD*DD + 255) / 256), 256, 0, stream>>>(Wvp,  wvp_bf, DD * DD);
  k_convert<<<dim3((DD*DD + 255) / 256), 256, 0, stream>>>(Wv,   wv_bf,  DD * DD);
  k_convert<<<dim3((H4*H4 + 255) / 256), 256, 0, stream>>>(Wg,   wg_bf,  H4 * H4);
  k_convert<<<dim3((TH*H4 + 255) / 256), 256, 0, stream>>>(W_ih, wih_bf, TH * H4);
  k_convert<<<dim3((TH*DD + 255) / 256), 256, 0, stream>>>(W_hh, whh_bf, TH * DD);

  // a = question @ Wuq^T  (M = B*Q = 4096, K = N = 512)
  k_gemm<<<dim3(DD / 64, (BB * QQ) / 16), 128, 0, stream>>>(
      question, (long)DD, wuq_bf, nullptr, a_ws, DD, DD, 0);

  // ---- sequential scan over timesteps ----
  for (int t = 0; t < PP; ++t) {
    const float* hprev = (t == 0) ? h0_ws : (out + (size_t)(t - 1) * DD);
    long hlda = (t == 0) ? (long)DD : (long)PP * DD;

    // add = p_t @ Wup^T + h @ Wvp^T   (M=64, K=N=512, fused dual-A GEMM)
    k_gemm2<<<dim3(DD / 64, BB / 16), 128, 0, stream>>>(
        passage + (size_t)t * DD, (long)PP * DD, wup_bf, DD,
        hprev, hlda, wvp_bf, DD, add_ws, DD);

    // s = tanh(a + add) @ Wv^T   (the big GEMM, TDM + LDS fused)
    k_attn<<<dim3(2, BB), 256, 0, stream>>>(a_ws, add_ws, wv_bf, s_ws);

    // batch-axis softmax + context vector
    k_sm_stats<<<dim3((QQ * DD + 255) / 256), 256, 0, stream>>>(s_ws, m_ws, z_ws);
    k_ctx<<<dim3((BB * DD + 255) / 256), 256, 0, stream>>>(s_ws, question, m_ws, z_ws, ct_ws);

    // pa = [p_t, c_t]; gate; x = sigmoid(pa@Wg^T) * pa
    k_pa<<<dim3((BB * H4 + 255) / 256), 256, 0, stream>>>(passage, ct_ws, pa_ws, t);
    k_gemm<<<dim3(H4 / 64, BB / 16), 128, 0, stream>>>(
        pa_ws, (long)H4, wg_bf, nullptr, gt_ws, H4, H4, 0);
    k_gate<<<dim3((BB * H4 + 255) / 256), 256, 0, stream>>>(gt_ws, pa_ws, x_ws);

    // GRU gates: gi = x@W_ih^T + b_ih ; gh = h@W_hh^T + b_hh
    k_gemm<<<dim3(TH / 64, BB / 16), 128, 0, stream>>>(
        x_ws, (long)H4, wih_bf, b_ih, gi_ws, TH, H4, 0);
    k_gemm<<<dim3(TH / 64, BB / 16), 128, 0, stream>>>(
        hprev, hlda, whh_bf, b_hh, gh_ws, TH, DD, 0);

    // h_t -> out[:, t, :]
    k_gru<<<dim3((BB * DD + 255) / 256), 256, 0, stream>>>(gi_ws, gh_ws, hprev, hlda, out, t);
  }
}
